// FBRNN_encoder_86904368267681
// MI455X (gfx1250) — matrix operand order repaired and checked by reading, not compiled, not run on hardware
//
#include <hip/hip_runtime.h>
#include <hip/hip_bf16.h>
#include <math.h>

typedef __attribute__((ext_vector_type(16))) __bf16 v16bf;
typedef __attribute__((ext_vector_type(8)))  float  v8f;

namespace {
constexpr int SEQ    = 2048;
constexpr int HID    = 512;
constexpr int LAYERS = 4;
constexpr int ATT_H  = 256;
constexpr int NWG    = 64;    // persistent workgroups (one per WGP)
constexpr int TPB    = 128;   // 4 waves of 32
constexpr int UPW    = 8;     // hidden units owned per WG (512/64)
// GRU A-fragments per WG: 4 layers * 2 matrices * 2 row-tiles * 16 k-chunks * (32 lanes*16 bf16)
constexpr int GRU_FRAG = LAYERS * 2 * 2 * 16 * 512;  // 131072 bf16 = 256 KB
// Attention A-fragments per WG: 1 row-tile (16 cols of W_i^T) * 16 k-chunks * 512
constexpr int ATT_FRAG = 16 * 512;                   // 8192 bf16 = 16 KB
}

typedef int v4i_ __attribute__((vector_size(16)));

#if __has_builtin(__builtin_amdgcn_global_load_async_to_lds_b128) && \
    __has_builtin(__builtin_amdgcn_s_wait_asynccnt)
#define USE_ASYNC_LDS 1
#else
#define USE_ASYNC_LDS 0
#endif

#if __has_builtin(__builtin_amdgcn_tensor_load_to_lds) && \
    __has_builtin(__builtin_amdgcn_s_wait_tensorcnt)
#define USE_TDM 1
typedef unsigned int u32x4 __attribute__((vector_size(16)));
typedef int          i32x8 __attribute__((vector_size(32)));
typedef int          i32x4 __attribute__((vector_size(16)));
#else
#define USE_TDM 0
#endif

// ---------------------------------------------------------------------------
// Pre-pass: swizzle GRU weights (f32 -> bf16) into WMMA A-fragment order.
// 16-bit A 16x32 layout: lane<16 holds row M=lane, K = {0..7}U{16..23};
// lane>=16 holds M=lane-16, K = {8..15}U{24..31}; element e -> VGPR e/2 half e%2.
// Row-tile packing per (layer, matrix): tile0 = [i_r j0..7, i_z j0..7],
// tile1 = [i_n j0..7, zero-pad].
// ---------------------------------------------------------------------------
__global__ void swz_gru(const float* __restrict__ w_ih,
                        const float* __restrict__ w_hh,
                        __bf16* __restrict__ out)
{
    int idx  = blockIdx.x * blockDim.x + threadIdx.x;   // < NWG*GRU_FRAG
    int e    =  idx        & 15;
    int lane = (idx >> 4)  & 31;
    int c    = (idx >> 9)  & 15;
    int tl   = (idx >> 13) & 1;
    int m    = (idx >> 14) & 1;
    int l    = (idx >> 15) & 3;
    int g    =  idx >> 17;
    int mm   = lane & 15;
    int half = lane >> 4;
    int koff = ((e >> 3) << 4) + (e & 7) + (half << 3);
    int k    = c * 32 + koff;
    float v  = 0.f;
    if (!(tl == 1 && mm >= 8)) {
        int gate = (tl == 0) ? (mm >> 3) : 2;
        int j    = g * UPW + (mm & 7);
        const float* w = m ? w_hh : w_ih;
        v = w[((size_t)l * 3 * HID + (size_t)gate * HID + j) * HID + k];
    }
    out[idx] = (__bf16)v;
}

// Attention: WG g -> head i=g/16, column-tile ct=g%16. A[M=c_local][K=j] =
// att_W[i][j][ct*16+c_local].
__global__ void swz_att(const float* __restrict__ att_W, __bf16* __restrict__ out)
{
    int idx  = blockIdx.x * blockDim.x + threadIdx.x;   // < NWG*ATT_FRAG
    int e    =  idx       & 15;
    int lane = (idx >> 4) & 31;
    int c    = (idx >> 9) & 15;
    int g    =  idx >> 13;
    int i    = g >> 4, ct = g & 15;
    int mm   = lane & 15;
    int half = lane >> 4;
    int koff = ((e >> 3) << 4) + (e & 7) + (half << 3);
    int j    = c * 32 + koff;
    out[idx] = (__bf16)att_W[((size_t)i * HID + j) * ATT_H + ct * 16 + mm];
}

__global__ void init_ws(float* ebuf, int* syncv)
{
    int t = threadIdx.x;
    if (t < 32) ebuf[t] = 0.f;
    if (t < 48) syncv[t] = 0;
}

__device__ __forceinline__ float sigmoidf_(float x)
{
    return __builtin_amdgcn_rcpf(1.f + __expf(-x));
}

__device__ __forceinline__ unsigned lds_offset_of(const void* p)
{
    return (unsigned)(unsigned long long)
        (__attribute__((address_space(3))) const char*)(const char*)p;
}

__device__ __forceinline__ void grid_sync(int* cnt, int* gen)
{
    __threadfence();
    __syncthreads();
    if (threadIdx.x == 0) {
        int old = __hip_atomic_load(gen, __ATOMIC_RELAXED, __HIP_MEMORY_SCOPE_AGENT);
        int arr = __hip_atomic_fetch_add(cnt, 1, __ATOMIC_ACQ_REL, __HIP_MEMORY_SCOPE_AGENT);
        if (arr == NWG - 1) {
            __hip_atomic_store(cnt, 0, __ATOMIC_RELAXED, __HIP_MEMORY_SCOPE_AGENT);
            __hip_atomic_fetch_add(gen, 1, __ATOMIC_RELEASE, __HIP_MEMORY_SCOPE_AGENT);
        } else {
            while (__hip_atomic_load(gen, __ATOMIC_ACQUIRE, __HIP_MEMORY_SCOPE_AGENT) == old)
                __builtin_amdgcn_s_sleep(1);
        }
    }
    __syncthreads();
}

// ---------------------------------------------------------------------------
// Persistent recurrent kernel: 64 WGs, LDS-resident bf16 weights (staged via
// TDM + async-to-LDS), WMMA matvecs with 4 interleaved accumulator chains,
// 5 global barriers per timestep.
// ---------------------------------------------------------------------------
__global__ __launch_bounds__(TPB) void rnn_persistent(
    const int*    __restrict__ batch,
    const float*  __restrict__ emb,
    const float*  __restrict__ b_ih,
    const float*  __restrict__ b_hh,
    const float*  __restrict__ att_v,
    const __bf16* __restrict__ gfrags,
    const __bf16* __restrict__ afrags,
    float*        __restrict__ xbuf,     // [LAYERS][HID] broadcast of layer outputs
    float*        __restrict__ ebuf,     // [2][16] ping-pong attention scores
    int*          __restrict__ syncv,    // [0]=cnt, [32]=gen
    float*        __restrict__ out)      // [SEQ][HID]
{
    __shared__ __align__(16) __bf16 wfrag[GRU_FRAG];     // 256 KB
    __shared__ __align__(16) __bf16 afrag[ATT_FRAG];     // 16 KB
    __shared__ float stackloc[LAYERS][HID];              // 8 KB
    __shared__ float hloc[LAYERS][HID];                  // 8 KB
    __shared__ float xloc[HID];                          // 2 KB
    __shared__ float gres[2][32];
    __shared__ float epart[8];
    __shared__ float swt[4][4];
    __shared__ float bihloc[96], bhhloc[96];
    __shared__ float attvloc[16];

    const int g    = blockIdx.x;
    const int tid  = threadIdx.x;
    const int lane = tid & 31;
    const int wave = tid >> 5;
    const int wm   = wave >> 1;   // 0 = W_ih, 1 = W_hh chain
    const int wtl  = wave & 1;    // row tile
    const int ai   = g >> 4;      // attention head
    const int act  = g & 15;      // attention column tile

    int* cnt = syncv;
    int* gen = syncv + 32;

    // ---- stage GRU weights into LDS via Tensor Data Mover (one 64KB quarter
    //      per wave): D# group0 = {count=1, lds_addr, global_addr, type=2},
    //      group1 = {data_size=8B, dim0=tile0=8192, dim1=tile1=1, stride0=8192}
#if USE_TDM
    {
        const unsigned ldsb = lds_offset_of(wfrag) + (unsigned)wave * 65536u;
        const unsigned long long ga =
            (unsigned long long)(const void*)(gfrags + (size_t)g * GRU_FRAG)
            + (unsigned long long)wave * 65536ull;
        const unsigned NE = 65536u / 8u;   // 8192 8-byte elements
        u32x4 d0 = { 1u,                                     // count=1, user mode
                     ldsb,                                   // lds_addr (bytes)
                     (unsigned)(ga & 0xFFFFFFFFull),         // global_addr lo
                     (unsigned)((ga >> 32) & 0x1FFFFFFull)   // global_addr hi
                       | (2u << 30) };                       // type=2 ("image")
        i32x8 d1 = { (int)(3u << 16),                 // data_size = 8 bytes
                     (int)((NE & 0xFFFFu) << 16),     // tensor_dim0[15:0]
                     (int)((NE >> 16) | (1u << 16)),  // tensor_dim0 hi | tensor_dim1=1
                     (int)((NE & 0xFFFFu) << 16),     // tensor_dim1 hi=0 | tile_dim0
                     1,                               // tile_dim1=1, tile_dim2=0
                     (int)NE,                         // tensor_dim0_stride lo
                     0, 0 };
        i32x4 dz4 = { 0, 0, 0, 0 };
        i32x8 dz8 = { 0, 0, 0, 0, 0, 0, 0, 0 };
        __builtin_amdgcn_tensor_load_to_lds(d0, d1, dz4, dz4, dz8, 0);
    }
#else
    {
        const uint4* s = (const uint4*)(gfrags + (size_t)g * GRU_FRAG);
        uint4* d = (uint4*)wfrag;
        for (int i = tid; i < GRU_FRAG / 8; i += TPB) d[i] = s[i];
    }
#endif
    // ---- stage attention weights via async global->LDS (ASYNCcnt path) ----
#if USE_ASYNC_LDS
    {
        char* asrc = (char*)(afrags + (size_t)g * ATT_FRAG);
        char* adst = (char*)afrag;
        for (int i = tid * 16; i < ATT_FRAG * 2; i += TPB * 16)
            __builtin_amdgcn_global_load_async_to_lds_b128(
                (__attribute__((address_space(1))) v4i_*)(asrc + i),
                (__attribute__((address_space(3))) v4i_*)(adst + i), 0, 0);
    }
#else
    {
        const uint4* s2 = (const uint4*)(afrags + (size_t)g * ATT_FRAG);
        uint4* d2 = (uint4*)afrag;
        for (int i = tid; i < ATT_FRAG / 8; i += TPB) d2[i] = s2[i];
    }
#endif
    for (int i = tid; i < 96; i += TPB) {
        int l = i / 24, r = i % 24, gate = r >> 3, j = r & 7;
        size_t o = (size_t)l * 3 * HID + (size_t)gate * HID + g * UPW + j;
        bihloc[i] = b_ih[o];
        bhhloc[i] = b_hh[o];
    }
    if (tid < 16) attvloc[tid] = att_v[(size_t)ai * ATT_H + act * 16 + tid];
    for (int i = tid; i < LAYERS * HID; i += TPB) (&hloc[0][0])[i] = 0.f;  // h0 = 0
#if USE_TDM
    __builtin_amdgcn_s_wait_tensorcnt(0);
#endif
#if USE_ASYNC_LDS
    __builtin_amdgcn_s_wait_asynccnt(0);
#endif
    __syncthreads();

    for (int t = 0; t < SEQ; ++t) {
        const int tok = batch[t];
        if (tid == 0 && t + 1 < SEQ)
            __builtin_prefetch(emb + (size_t)batch[t + 1] * HID, 0, 0);  // global_prefetch
        for (int i = tid; i < HID; i += TPB) xloc[i] = emb[(size_t)tok * HID + i];
        __syncthreads();

        // ---------------- GRU layers (serial, x broadcast via L2) ------------
        for (int l = 0; l < LAYERS; ++l) {
            const float*  bsrc = wm ? hloc[l] : xloc;
            const __bf16* wa   = wfrag + (size_t)(((l * 2 + wm) * 2 + wtl) * 16) * 512;
            v8f ac0 = {}, ac1 = {}, ac2 = {}, ac3 = {};
            // B fragments hoisted out of the loop: dead-zero lanes stay
            // resident in VGPRs; only the column-0 lanes (0, 16) are rewritten per chunk.
            v16bf b0 = {}, b1 = {}, b2 = {}, b3 = {};
            #pragma unroll
            for (int c = 0; c < 16; c += 4) {
                v16bf a0 = *(const v16bf*)(wa + (c + 0) * 512 + lane * 16);
                v16bf a1 = *(const v16bf*)(wa + (c + 1) * 512 + lane * 16);
                v16bf a2 = *(const v16bf*)(wa + (c + 2) * 512 + lane * 16);
                v16bf a3 = *(const v16bf*)(wa + (c + 3) * 512 + lane * 16);
                if ((lane & 15) == 0) {                      // only B column 0 live
                    #pragma unroll
                    for (int s = 0; s < 16; ++s) {
                        const int kk = (lane >> 4) * 8 + (s & 7) + ((s >> 3) << 4);
                        b0[s] = (__bf16)bsrc[(c + 0) * 32 + kk];
                        b1[s] = (__bf16)bsrc[(c + 1) * 32 + kk];
                        b2[s] = (__bf16)bsrc[(c + 2) * 32 + kk];
                        b3[s] = (__bf16)bsrc[(c + 3) * 32 + kk];
                    }
                }
                ac0 = __builtin_amdgcn_wmma_f32_16x16x32_bf16(false, a0, false, b0, (short)0, ac0, false, false);
                ac1 = __builtin_amdgcn_wmma_f32_16x16x32_bf16(false, a1, false, b1, (short)0, ac1, false, false);
                ac2 = __builtin_amdgcn_wmma_f32_16x16x32_bf16(false, a2, false, b2, (short)0, ac2, false, false);
                ac3 = __builtin_amdgcn_wmma_f32_16x16x32_bf16(false, a3, false, b3, (short)0, ac3, false, false);
            }
            v8f acc = (ac0 + ac1) + (ac2 + ac3);
            if (lane == 0 || lane == 16) {                   // D column 0 lives here
                const int rb = (lane >> 4) * 8;
                #pragma unroll
                for (int r = 0; r < 8; ++r) gres[wm][wtl * 16 + rb + r] = acc[r];
            }
            __syncthreads();
            if (tid < UPW) {                                  // GRU elementwise
                const int j = tid, J = g * UPW + j;
                const float i_r = gres[0][j]      + bihloc[l * 24 + 0 + j];
                const float i_z = gres[0][8 + j]  + bihloc[l * 24 + 8 + j];
                const float i_n = gres[0][16 + j] + bihloc[l * 24 + 16 + j];
                const float h_r = gres[1][j]      + bhhloc[l * 24 + 0 + j];
                const float h_z = gres[1][8 + j]  + bhhloc[l * 24 + 8 + j];
                const float h_n = gres[1][16 + j] + bhhloc[l * 24 + 16 + j];
                const float r = sigmoidf_(i_r + h_r);
                const float z = sigmoidf_(i_z + h_z);
                const float n = tanhf(i_n + r * h_n);
                const float xn = (1.f - z) * n + z * hloc[l][J];
                xbuf[l * HID + J] = xn;
                if (l == LAYERS - 1) out[(size_t)t * HID + J] = xn;  // combined[3]==x4
            }
            grid_sync(cnt, gen);
            for (int i = tid; i < HID; i += TPB) {
                float v = xbuf[l * HID + i];
                stackloc[l][i] = v;
                if (l < LAYERS - 1) xloc[i] = v;
            }
            __syncthreads();
        }

        // ------------- attention scores e[i][k] = v_i . tanh(W_i^T x_k) ------
        if (wave == 0) {
            v8f ac0 = {}, ac1 = {}, ac2 = {}, ac3 = {};
            v16bf b0 = {}, b1 = {}, b2 = {}, b3 = {};
            #pragma unroll
            for (int c = 0; c < 16; c += 4) {
                v16bf a0 = *(const v16bf*)(afrag + (c + 0) * 512 + lane * 16);
                v16bf a1 = *(const v16bf*)(afrag + (c + 1) * 512 + lane * 16);
                v16bf a2 = *(const v16bf*)(afrag + (c + 2) * 512 + lane * 16);
                v16bf a3 = *(const v16bf*)(afrag + (c + 3) * 512 + lane * 16);
                #pragma unroll
                for (int s = 0; s < 16; ++s) {               // 4 of 16 B columns live
                    if ((lane & 15) < 4) {
                        const int kk = (lane >> 4) * 8 + (s & 7) + ((s >> 3) << 4);
                        b0[s] = (__bf16)stackloc[lane & 15][(c + 0) * 32 + kk];
                        b1[s] = (__bf16)stackloc[lane & 15][(c + 1) * 32 + kk];
                        b2[s] = (__bf16)stackloc[lane & 15][(c + 2) * 32 + kk];
                        b3[s] = (__bf16)stackloc[lane & 15][(c + 3) * 32 + kk];
                    }
                }
                ac0 = __builtin_amdgcn_wmma_f32_16x16x32_bf16(false, a0, false, b0, (short)0, ac0, false, false);
                ac1 = __builtin_amdgcn_wmma_f32_16x16x32_bf16(false, a1, false, b1, (short)0, ac1, false, false);
                ac2 = __builtin_amdgcn_wmma_f32_16x16x32_bf16(false, a2, false, b2, (short)0, ac2, false, false);
                ac3 = __builtin_amdgcn_wmma_f32_16x16x32_bf16(false, a3, false, b3, (short)0, ac3, false, false);
            }
            v8f acc = (ac0 + ac1) + (ac2 + ac3);
            if ((lane & 15) < 4) {            // lanes 0-3 / 16-19 hold columns k=0..3
                const int k  = lane & 3;
                const int mb = (lane >> 4) * 8;
                float s = 0.f;
                #pragma unroll
                for (int r = 0; r < 8; ++r) s += tanhf(acc[r]) * attvloc[mb + r];
                epart[(lane >> 4) * 4 + k] = s;
            }
        }
        if (g == 0 && tid < 16) ebuf[(1 - (t & 1)) * 16 + tid] = 0.f;  // re-arm other buf
        __syncthreads();
        if (tid < 4)
            atomicAdd(&ebuf[(t & 1) * 16 + ai * 4 + tid], epart[tid] + epart[4 + tid]);
        grid_sync(cnt, gen);

        // ------------- softmax + combined h (computed redundantly per WG) ----
        if (tid < 4) {
            const int i = tid;
            volatile const float* e = ebuf + (t & 1) * 16;
            float mx = -1e30f;
            for (int k = i; k < 4; ++k) mx = fmaxf(mx, e[i * 4 + k]);
            float w[4], s = 0.f;
            for (int k = 0; k < 4; ++k) {
                float v = (k >= i) ? __expf(e[i * 4 + k] - mx) : 0.f;
                w[k] = v; s += v;
            }
            const float rs = __builtin_amdgcn_rcpf(s);
            for (int k = 0; k < 4; ++k) swt[i][k] = w[k] * rs;
        }
        __syncthreads();
        for (int idx = tid; idx < HID; idx += TPB) {
            #pragma unroll
            for (int i = 0; i < 4; ++i) {
                float h = 0.f;
                #pragma unroll
                for (int k = 0; k < 4; ++k) h += swt[i][k] * stackloc[k][idx];
                hloc[i][idx] = h;
            }
        }
        __syncthreads();
    }
}

// ---------------------------------------------------------------------------
extern "C" void kernel_launch(void* const* d_in, const int* in_sizes, int n_in,
                              void* d_out, int out_size, void* d_ws, size_t ws_size,
                              hipStream_t stream)
{
    (void)in_sizes; (void)n_in; (void)out_size; (void)ws_size;
    const int*   batch = (const int*)  d_in[0];
    const float* emb   = (const float*)d_in[1];
    const float* w_ih  = (const float*)d_in[2];
    const float* w_hh  = (const float*)d_in[3];
    const float* b_ih  = (const float*)d_in[4];
    const float* b_hh  = (const float*)d_in[5];
    const float* att_W = (const float*)d_in[6];
    const float* att_v = (const float*)d_in[7];
    float* out = (float*)d_out;

    char*   ws    = (char*)d_ws;
    __bf16* gfr   = (__bf16*)ws;                                          // 16 MB
    __bf16* afr   = (__bf16*)(ws + (size_t)NWG * GRU_FRAG * 2);           // 1 MB
    float*  xbuf  = (float*)(ws + (size_t)NWG * (GRU_FRAG + ATT_FRAG) * 2);
    float*  ebuf  = xbuf + LAYERS * HID;
    int*    syncv = (int*)(ebuf + 32);

    hipLaunchKernelGGL(swz_gru, dim3(NWG * GRU_FRAG / 256), dim3(256), 0, stream,
                       w_ih, w_hh, gfr);
    hipLaunchKernelGGL(swz_att, dim3(NWG * ATT_FRAG / 256), dim3(256), 0, stream,
                       att_W, afr);
    hipLaunchKernelGGL(init_ws, dim3(1), dim3(64), 0, stream, ebuf, syncv);
    hipLaunchKernelGGL(rnn_persistent, dim3(NWG), dim3(TPB), 0, stream,
                       batch, emb, b_ih, b_hh, att_v, gfr, afr, xbuf, ebuf, syncv, out);
}
